// PairFormerBlock_88115549045153
// MI455X (gfx1250) — matrix operand order repaired and checked by reading, not compiled, not run on hardware
//
#include <hip/hip_runtime.h>
#include <stdint.h>

typedef __attribute__((ext_vector_type(16))) __bf16 v16bf;
typedef __attribute__((ext_vector_type(8)))  float  v8f;

#define WMMA_BF16(A_, B_, C_) \
  __builtin_amdgcn_wmma_f32_16x16x32_bf16(false, (A_), false, (B_), (short)0, (C_), false, false)

__device__ __forceinline__ float sigm(float x) { return 1.f / (1.f + __expf(-x)); }

// A-fragment K permutation: wave32 WMMA A layout wants per-lane K order
// {kbase..kbase+7, 16+kbase..16+kbase+7}. Swapping bits 3<->4 of the K index
// makes each lane's 16 values contiguous in LDS. Involution (self-inverse).
__device__ __forceinline__ int kperm(int c) {
  return (c & 7) | ((c >> 1) & 8) | ((c << 1) & 16);
}

// ---------------------------------------------------------------------------
// Generic strided batched GEMM: C[m,n] = sum_k A[m,k]*B[k,n] (+bias[n])
// bf16 operands staged in LDS in fragment order, f32 accumulate via
// v_wmma_f32_16x16x32_bf16. Block tile 64x64, 8 waves, 16x32 strip per wave.
// ---------------------------------------------------------------------------
struct GemmDesc {
  const float* A; const float* B; const float* bias; float* C;
  int M, N, K;
  long sAr, sAk, sBk, sBn, sCr, sCc;
  int nInner;                       // batch z = outer*nInner + inner
  long aO, aI, bO, bI, cO, cI;      // batch strides
};

__global__ __launch_bounds__(256) void k_gemm(GemmDesc d) {
  int ob = blockIdx.z / d.nInner, ib = blockIdx.z % d.nInner;
  const float* A = d.A + (long)ob * d.aO + (long)ib * d.aI;
  const float* B = d.B + (long)ob * d.bO + (long)ib * d.bI;
  float*       C = d.C + (long)ob * d.cO + (long)ib * d.cI;
  int m0 = blockIdx.y * 64, n0 = blockIdx.x * 64;

  __shared__ v16bf AsV[128];        // A tile 64x32, rows in natural k order (frag order)
  __shared__ v16bf BtV[128];        // B tile transposed [n][k] 64x32 (frag order)
  __bf16* As = (__bf16*)AsV;
  __bf16* Bt = (__bf16*)BtV;

  int tid = threadIdx.x, lane = tid & 31, wave = tid >> 5;
  int wr = wave & 3, wc = wave >> 2;          // 4 row-tiles x 2 col-strips
  int half = lane >> 4, l16 = lane & 15;

  bool fullM = (m0 + 64 <= d.M);
  bool fullN = (n0 + 64 <= d.N);

  v8f acc0 = {}, acc1 = {};
  for (int k0 = 0; k0 < d.K; k0 += 32) {
    bool fullK = (k0 + 32 <= d.K);
    // ---- A tile ----
    if (fullM && fullK) {
      if (d.sAk == 1) {
        for (int t = tid; t < 512; t += 256) {          // float4 fast path
          int r = t >> 3, c = (t & 7) * 4;
          const float4 f = *(const float4*)(A + (long)(m0 + r) * d.sAr + (k0 + c));
          int p = r * 32 + c;                           // 4 dests contiguous
          As[p + 0] = (__bf16)f.x; As[p + 1] = (__bf16)f.y;
          As[p + 2] = (__bf16)f.z; As[p + 3] = (__bf16)f.w;
        }
      } else {
        for (int t = tid; t < 2048; t += 256) {
          int r = t >> 5, c = t & 31;
          As[r * 32 + c] =
              (__bf16)A[(long)(m0 + r) * d.sAr + (long)(k0 + c) * d.sAk];
        }
      }
    } else {
      for (int t = tid; t < 2048; t += 256) {
        int r = t >> 5, c = t & 31;
        int gm = m0 + r, gk = k0 + c;
        float v = (gm < d.M && gk < d.K) ? A[(long)gm * d.sAr + (long)gk * d.sAk] : 0.f;
        As[r * 32 + c] = (__bf16)v;
      }
    }
    // ---- B tile (store transposed [n][k]) ----
    if (fullN && fullK) {
      if (d.sBn == 1) {
        for (int t = tid; t < 512; t += 256) {          // float4 fast path
          int k = t >> 4, n = (t & 15) * 4;
          const float4 f = *(const float4*)(B + (long)(k0 + k) * d.sBk + (n0 + n));
          Bt[(n + 0) * 32 + k] = (__bf16)f.x; Bt[(n + 1) * 32 + k] = (__bf16)f.y;
          Bt[(n + 2) * 32 + k] = (__bf16)f.z; Bt[(n + 3) * 32 + k] = (__bf16)f.w;
        }
      } else {
        for (int t = tid; t < 2048; t += 256) {
          int k = t >> 6, n = t & 63;
          Bt[n * 32 + k] =
              (__bf16)B[(long)(k0 + k) * d.sBk + (long)(n0 + n) * d.sBn];
        }
      }
    } else {
      for (int t = tid; t < 2048; t += 256) {
        int k = t >> 6, n = t & 63;
        int gk = k0 + k, gn = n0 + n;
        float v = (gk < d.K && gn < d.N) ? B[(long)gk * d.sBk + (long)gn * d.sBn] : 0.f;
        Bt[n * 32 + k] = (__bf16)v;
      }
    }
    if (k0 + 32 < d.K)                                  // speculative prefetch
      __builtin_prefetch(A + (long)(m0 + (tid & 63)) * d.sAr + (long)(k0 + 32) * d.sAk, 0, 1);
    __syncthreads();

    v16bf af = AsV[(wr * 16 + l16) * 2 + half];         // 1 x 32B LDS load
    v16bf b0 = BtV[(wc * 32 + l16) * 2 + half];
    v16bf b1 = BtV[(wc * 32 + 16 + l16) * 2 + half];
    acc0 = WMMA_BF16(af, b0, acc0);
    acc1 = WMMA_BF16(af, b1, acc1);
    __syncthreads();
  }
#pragma unroll
  for (int g = 0; g < 8; ++g) {                         // C: vgpr g -> rows g / g+8
    int gm = m0 + wr * 16 + g + half * 8;
    if (gm >= d.M) continue;
    int gn0 = n0 + wc * 32 + l16;
    if (gn0 < d.N) {
      float b0 = d.bias ? d.bias[gn0] : 0.f;
      C[(long)gm * d.sCr + (long)gn0 * d.sCc] = acc0[g] + b0;
    }
    int gn1 = gn0 + 16;
    if (gn1 < d.N) {
      float b1 = d.bias ? d.bias[gn1] : 0.f;
      C[(long)gm * d.sCr + (long)gn1 * d.sCc] = acc1[g] + b1;
    }
  }
}

// ---------------------------------------------------------------------------
// LayerNorm over last dim D, one wave per row, optional affine.
// ---------------------------------------------------------------------------
__global__ __launch_bounds__(256) void k_ln(const float* x, float* y,
                                            const float* g, const float* b,
                                            int rows, int D) {
  int wv = threadIdx.x >> 5, lane = threadIdx.x & 31;
  int row = blockIdx.x * 8 + wv;
  if (row >= rows) return;
  const float* xr = x + (long)row * D;
  float s = 0.f, ss = 0.f;
  for (int i = lane; i < D; i += 32) { float v = xr[i]; s += v; ss += v * v; }
  for (int m = 16; m >= 1; m >>= 1) { s += __shfl_xor(s, m, 32); ss += __shfl_xor(ss, m, 32); }
  float mean = s / D;
  float inv = rsqrtf(ss / D - mean * mean + 1e-5f);
  float* yr = y + (long)row * D;
  for (int i = lane; i < D; i += 32) {
    float v = (xr[i] - mean) * inv;
    if (g) v = v * g[i] + b[i];
    yr[i] = v;
  }
}

// adaLN: y = ln(s) * (1 + scale[b]) + shift[b], ss = [B, 2D]
__global__ __launch_bounds__(256) void k_adaln(const float* s, const float* ss, float* y,
                                               int Bb, int Nn, int D) {
  int wv = threadIdx.x >> 5, lane = threadIdx.x & 31;
  int row = blockIdx.x * 8 + wv;
  if (row >= Bb * Nn) return;
  int b = row / Nn;
  const float* xr = s + (long)row * D;
  float su = 0.f, sq = 0.f;
  for (int i = lane; i < D; i += 32) { float v = xr[i]; su += v; sq += v * v; }
  for (int m = 16; m >= 1; m >>= 1) { su += __shfl_xor(su, m, 32); sq += __shfl_xor(sq, m, 32); }
  float mean = su / D;
  float inv = rsqrtf(sq / D - mean * mean + 1e-5f);
  const float* sc = ss + (long)b * 2 * D;
  float* yr = y + (long)row * D;
  for (int i = lane; i < D; i += 32) {
    float v = (xr[i] - mean) * inv;
    yr[i] = v * (1.f + sc[i]) + sc[D + i];
  }
}

// zm[b,j,d] = mean_i zi[b,i,j,d]
__global__ void k_mean_i(const float* zi, float* zm, int B, int N, int D) {
  long total = (long)B * N * D;
  long idx = (long)blockIdx.x * blockDim.x + threadIdx.x;
  if (idx >= total) return;
  int d = (int)(idx % D); long r = idx / D; int j = (int)(r % N); int b = (int)(r / N);
  const float* p = zi + ((long)b * N * N + j) * D + d;
  float acc = 0.f;
  for (int i = 0; i < N; ++i) acc += p[(long)i * N * D];
  zm[idx] = acc / (float)N;
}

// p[b,i,j,d] *= sigmoid(g[idx]) * (mask[b,i] && mask[b,j])
__global__ void k_gate_mask(float* p, const float* g, const int* mask, int B, int N, int D) {
  long total = (long)B * N * N * D;
  long idx = (long)blockIdx.x * blockDim.x + threadIdx.x;
  if (idx >= total) return;
  long r = idx / D; int j = (int)(r % N); r /= N; int i = (int)(r % N); int b = (int)(r / N);
  float m = (mask[b * N + i] && mask[b * N + j]) ? 1.f : 0.f;
  p[idx] = p[idx] * sigm(g[idx]) * m;
}

// dst[b,i,j,d] += sigmoid(graw[src]) * x[src]; src transposed in (i,j) if requested
__global__ void k_resid_gate(float* dst, const float* graw, const float* x,
                             int B, int N1, int N2, int D, int transpose) {
  long total = (long)B * N1 * N2 * D;
  long idx = (long)blockIdx.x * blockDim.x + threadIdx.x;
  if (idx >= total) return;
  long src = idx;
  if (transpose) {
    int dd = (int)(idx % D); long r = idx / D;
    int j = (int)(r % N2); r /= N2; int i = (int)(r % N1); int b = (int)(r / N1);
    src = (((long)b * N1 + j) * N2 + i) * D + dd;
  }
  dst[idx] += sigm(graw[src]) * x[src];
}

// zt[b,i,j,d] = z[b,j,i,d]
__global__ void k_transpose(const float* z, float* zt, int B, int N, int D) {
  long total = (long)B * N * N * D;
  long idx = (long)blockIdx.x * blockDim.x + threadIdx.x;
  if (idx >= total) return;
  int dd = (int)(idx % D); long r = idx / D;
  int j = (int)(r % N); r /= N; int i = (int)(r % N); int b = (int)(r / N);
  zt[idx] = z[(((long)b * N + j) * N + i) * D + dd];
}

// ---------------------------------------------------------------------------
// Fused attention: per block = one (b,i,h) x 64 queries (4 waves x 16 rows).
// QK^T -> scale -> +bias -> mask -> softmax (regs) -> restage S to LDS -> S@V.
// All operands staged in LDS in WMMA fragment order (one v16bf per lane).
// Nk fixed = 192. HD templated (16: K-dim zero-padded to 32; 32: exact).
// ---------------------------------------------------------------------------
struct AttnDesc {
  const float* Q; const float* K; const float* V; const float* bias;
  const int* mask; float* O;
  float scale;
  int nI, H, Nmask;
  long sB, sI; int sRow;                 // Q/K/V batch/row strides (shared layout)
  long bsB; int bsQ, bsK; int biasMode;  // 0: bias[b,q,h]; 1: bias[b,q,k,h]
  long osB, osI; int osQ;
};

template <int HD>
__global__ __launch_bounds__(128) void k_attn(AttnDesc d) {
  constexpr int NK = 192;
  constexpr int NKV = NK / 16;           // v16bf per 192-wide row
  int q0 = blockIdx.x * 64;
  int bz = blockIdx.y;
  int h = bz % d.H; int rem = bz / d.H; int i = rem % d.nI; int b = rem / d.nI;
  long base = (long)b * d.sB + (long)i * d.sI + (long)h * HD;
  const float* Qp = d.Q + base;
  const float* Kp = d.K + base;
  const float* Vp = d.V + base;
  float* Op = d.O + (long)b * d.osB + (long)i * d.osI + (long)h * HD;
  const float* biasP = d.bias + (long)b * d.bsB + h;
  const int* maskP = d.mask + (long)b * d.Nmask;

  __shared__ v16bf KsV[NK * 2];          // [key][dd 0..31], zero-pad if HD==16
  __shared__ v16bf QsV[64 * 2];          // [q][dd 0..31], zero-pad if HD==16
  __shared__ v16bf VtV[HD * NKV];        // [dd][key] transposed
  __shared__ v16bf SsV[4 * 16 * NKV];    // [wave][row][key], contiguous key order
  __bf16* Ks = (__bf16*)KsV;
  __bf16* Qs = (__bf16*)QsV;
  __bf16* Vt = (__bf16*)VtV;
  __bf16* Ss = (__bf16*)SsV;

  int tid = threadIdx.x, lane = tid & 31, wv = tid >> 5;
  int half = lane >> 4, l16 = lane & 15;

  for (int t = tid; t < NK * HD; t += 128) {
    int r = t / HD, c = t % HD;
    Ks[r * 32 + c] = (__bf16)Kp[(long)r * d.sRow + c];
    Vt[c * NK + r] = (__bf16)Vp[(long)r * d.sRow + c];
  }
  if constexpr (HD == 16) {
    for (int t = tid; t < NK * 16; t += 128) {
      int r = t >> 4, c = t & 15;
      Ks[r * 32 + 16 + c] = (__bf16)0.f;
    }
  }
  if constexpr (HD == 32) {
    for (int t = tid; t < 64 * 32; t += 128) {
      int r = t >> 5, p = t & 31;
      Qs[r * 32 + p] = (__bf16)Qp[(long)(q0 + r) * d.sRow + p];
    }
  } else {
    for (int t = tid; t < 64 * 16; t += 128) {
      int r = t >> 4, k = t & 15;
      Qs[r * 32 + k] = (__bf16)Qp[(long)(q0 + r) * d.sRow + k];
      Qs[r * 32 + 16 + k] = (__bf16)0.f;
    }
  }
  __syncthreads();

  v16bf qa = QsV[(wv * 16 + l16) * 2 + half];

  v8f sc[NKV];
#pragma unroll
  for (int t = 0; t < NKV; ++t) {
    v16bf kb = KsV[(t * 16 + l16) * 2 + half];
    v8f z = {};
    sc[t] = WMMA_BF16(qa, kb, z);
  }

#pragma unroll
  for (int g = 0; g < 8; ++g) {
    int qloc = g + half * 8;
    int qg = q0 + wv * 16 + qloc;
    float bq = (d.biasMode == 0) ? biasP[(long)qg * d.bsQ] : 0.f;
    float vals[NKV];
    float mx = -3.4e38f;
#pragma unroll
    for (int t = 0; t < NKV; ++t) {
      int kkey = t * 16 + l16;
      float v = sc[t][g] * d.scale + bq;
      if (d.biasMode == 1) v += biasP[(long)qg * d.bsQ + (long)kkey * d.bsK];
      if (!maskP[kkey]) v = -3.4e38f;
      vals[t] = v;
      mx = fmaxf(mx, v);
    }
    for (int m = 1; m < 16; m <<= 1) mx = fmaxf(mx, __shfl_xor(mx, m, 32));
    float sum = 0.f;
#pragma unroll
    for (int t = 0; t < NKV; ++t) { float e = __expf(vals[t] - mx); vals[t] = e; sum += e; }
    for (int m = 1; m < 16; m <<= 1) sum += __shfl_xor(sum, m, 32);
    float inv = 1.f / sum;
#pragma unroll
    for (int t = 0; t < NKV; ++t) {
      int kkey = t * 16 + l16;
      int pc = kkey;                                     // contiguous key order
      Ss[((long)(wv * 16 + qloc)) * NK + pc] = (__bf16)(vals[t] * inv);
    }
  }
  __syncthreads();

  v8f oacc[HD / 16];
#pragma unroll
  for (int c = 0; c < HD / 16; ++c) { v8f z = {}; oacc[c] = z; }
#pragma unroll
  for (int s0 = 0; s0 < NK; s0 += 32) {
    v16bf sa = SsV[(wv * 16 + l16) * NKV + (s0 >> 4) + half];
#pragma unroll
    for (int c = 0; c < HD / 16; ++c) {
      v16bf vb = VtV[(c * 16 + l16) * NKV + (s0 >> 4) + half];
      oacc[c] = WMMA_BF16(sa, vb, oacc[c]);
    }
  }
#pragma unroll
  for (int c = 0; c < HD / 16; ++c)
#pragma unroll
    for (int g = 0; g < 8; ++g) {
      int qg = q0 + wv * 16 + g + half * 8;
      Op[(long)qg * d.osQ + c * 16 + l16] = oacc[c][g];
    }
}

// ---------------------------------------------------------------------------
// Host orchestration
// ---------------------------------------------------------------------------
extern "C" void kernel_launch(void* const* d_in, const int* in_sizes, int n_in,
                              void* d_out, int out_size, void* d_ws, size_t ws_size,
                              hipStream_t stream) {
  (void)in_sizes; (void)n_in; (void)out_size; (void)ws_size;
  const int B = 2, N = 192, DS = 256, DZ = 64, DC = 256, H = 8, HZ = 4;
  const long ZE = (long)B * N * N * DZ;   // 4,718,592
  const long SN = (long)B * N * DS;       // 98,304
  const long BNN = (long)B * N * N;       // 73,728
  const long BN = (long)B * N;            // 384

  auto F = [&](int i) { return (const float*)d_in[i]; };
  const float* s_in = F(0);
  const float* z_in = F(1);
  const float* cond = F(2);
  const int* mask = (const int*)d_in[3];
  // Param flat order (dict insertion order, leaves):
  // 4 adaln.proj.w 5 .b | 6..15 attn q/k/v/gate/out (w,b) | 16,17 attn.zn g,b | 18 attn.zproj.w
  // 19..34 tri_mul_out | 35..50 tri_mul_in | 51..63 tri_attn_start | 64..76 tri_attn_end

  float* ws = (float*)d_ws;
  float* zcur = ws;
  float* zi = ws + ZE;
  float* t0 = ws + 2 * ZE;
  float* t1 = ws + 3 * ZE;
  float* t2 = ws + 4 * ZE;
  float* t3 = ws + 5 * ZE;
  float* t4 = ws + 6 * ZE;
  float* sm = ws + 7 * ZE;
  float* ssb = sm; sm += (long)B * 2 * DS;
  float* xs  = sm; sm += SN;
  float* sq  = sm; sm += SN;
  float* sk  = sm; sm += SN;
  float* sv  = sm; sm += SN;
  float* sg  = sm; sm += SN;
  float* so  = sm; sm += SN;
  float* sol = sm; sm += SN;
  float* pb  = sm; sm += BNN * H;
  float* zm  = sm; sm += BN * DZ;
  float* bv  = sm; sm += BN * HZ;

  auto gemm = [&](const float* A, const float* Bw, const float* bias, float* C,
                  int M, int Nn, int K) {
    GemmDesc gd{A, Bw, bias, C, M, Nn, K,
                (long)K, 1L, (long)Nn, 1L, (long)Nn, 1L,
                1, 0L, 0L, 0L, 0L, 0L, 0L};
    dim3 grid((Nn + 63) / 64, (M + 63) / 64, 1);
    k_gemm<<<grid, 256, 0, stream>>>(gd);
  };
  auto ln = [&](const float* x, float* y, const float* g, const float* b, long rows, int D) {
    k_ln<<<dim3((unsigned)((rows + 7) / 8)), 256, 0, stream>>>(x, y, g, b, (int)rows, D);
  };
  auto ewg = [&](long total) { return dim3((unsigned)((total + 255) / 256)); };

  hipMemcpyAsync(zcur, z_in, ZE * sizeof(float), hipMemcpyDeviceToDevice, stream);

  // ---- triangle multiplications ----
  auto tri_mul = [&](int p, bool outgoing) {
    ln(zcur, zi, F(p), F(p + 1), BNN, DZ);
    gemm(zi, F(p + 2), F(p + 3), t0, (int)BNN, DZ, DZ);   // lp
    gemm(zi, F(p + 4), F(p + 5), t4, (int)BNN, DZ, DZ);   // lg
    k_gate_mask<<<ewg(ZE), 256, 0, stream>>>(t0, t4, mask, B, N, DZ);
    gemm(zi, F(p + 6), F(p + 7), t1, (int)BNN, DZ, DZ);   // rp
    gemm(zi, F(p + 8), F(p + 9), t4, (int)BNN, DZ, DZ);   // rg
    k_gate_mask<<<ewg(ZE), 256, 0, stream>>>(t1, t4, mask, B, N, DZ);
    {
      long sAr = outgoing ? (long)N * DZ : (long)DZ;      // A[i,k] vs A[k,i]
      long sAk = outgoing ? (long)DZ : (long)N * DZ;
      GemmDesc gd{t0, t1, nullptr, t2, N, N, N,
                  sAr, sAk, (long)N * DZ, (long)DZ, (long)N * DZ, (long)DZ,
                  DZ, (long)N * N * DZ, 1L, (long)N * N * DZ, 1L, (long)N * N * DZ, 1L};
      dim3 grid((N + 63) / 64, (N + 63) / 64, B * DZ);
      k_gemm<<<grid, 256, 0, stream>>>(gd);
    }
    ln(t2, t2, F(p + 10), F(p + 11), BNN, DZ);            // onorm
    gemm(t2, F(p + 12), F(p + 13), t3, (int)BNN, DZ, DZ); // op
    gemm(zi, F(p + 14), F(p + 15), t4, (int)BNN, DZ, DZ); // og
    k_resid_gate<<<ewg(ZE), 256, 0, stream>>>(zcur, t4, t3, B, N, N, DZ, 0);
  };
  tri_mul(19, true);
  tri_mul(35, false);

  // ---- triangle attentions ----
  auto tri_attn = [&](int p, bool transposed) {
    const float* zsrc = zcur;
    if (transposed) {
      k_transpose<<<ewg(ZE), 256, 0, stream>>>(zcur, t2, B, N, DZ);
      zsrc = t2;
    }
    ln(zsrc, zi, F(p), F(p + 1), BNN, DZ);
    gemm(zi, F(p + 2), F(p + 3), t0, (int)BNN, DZ, DZ);   // q
    gemm(zi, F(p + 4), F(p + 5), t1, (int)BNN, DZ, DZ);   // k
    float* vbuf = transposed ? t3 : t2;
    gemm(zi, F(p + 6), F(p + 7), vbuf, (int)BNN, DZ, DZ); // v
    k_mean_i<<<ewg(BN * DZ), 256, 0, stream>>>(zi, zm, B, N, DZ);
    gemm(zm, F(p + 8), nullptr, bv, (int)BN, HZ, DZ);     // bias
    {
      AttnDesc ad;
      ad.Q = t0; ad.K = t1; ad.V = vbuf; ad.bias = bv; ad.mask = mask; ad.O = zi;
      ad.scale = 0.25f;                                   // 1/sqrt(16)
      ad.nI = N; ad.H = HZ; ad.Nmask = N;
      ad.sB = (long)N * N * DZ; ad.sI = (long)N * DZ; ad.sRow = DZ;
      ad.bsB = (long)N * HZ; ad.bsQ = HZ; ad.bsK = 0; ad.biasMode = 0;
      ad.osB = (long)N * N * DZ; ad.osI = (long)N * DZ; ad.osQ = DZ;
      k_attn<16><<<dim3(3, (unsigned)(B * N * HZ)), 128, 0, stream>>>(ad);
    }
    gemm(zsrc, F(p + 9), F(p + 10), t4, (int)BNN, DZ, DZ); // gate on (possibly transposed) z
    gemm(zi, F(p + 11), F(p + 12), t0, (int)BNN, DZ, DZ);  // out proj of o
    k_resid_gate<<<ewg(ZE), 256, 0, stream>>>(zcur, t4, t0, B, N, N, DZ, transposed ? 1 : 0);
  };
  tri_attn(51, false);
  tri_attn(64, true);

  // ---- adaLN + pair-biased attention on s ----
  gemm(cond, F(4), F(5), ssb, B, 2 * DS, DC);
  k_adaln<<<dim3((unsigned)((BN + 7) / 8)), 256, 0, stream>>>(s_in, ssb, xs, B, N, DS);
  gemm(xs, F(6), F(7), sq, (int)BN, DS, DS);
  gemm(xs, F(8), F(9), sk, (int)BN, DS, DS);
  gemm(xs, F(10), F(11), sv, (int)BN, DS, DS);
  ln(zcur, zi, F(16), F(17), BNN, DZ);                    // zn
  gemm(zi, F(18), nullptr, pb, (int)BNN, H, DZ);          // zproj -> [B,N,N,H]
  {
    AttnDesc ad;
    ad.Q = sq; ad.K = sk; ad.V = sv; ad.bias = pb; ad.mask = mask; ad.O = so;
    ad.scale = 0.17677669529663687f;                      // 1/sqrt(32)
    ad.nI = 1; ad.H = H; ad.Nmask = N;
    ad.sB = (long)N * DS; ad.sI = 0; ad.sRow = DS;
    ad.bsB = (long)N * N * H; ad.bsQ = N * H; ad.bsK = H; ad.biasMode = 1;
    ad.osB = (long)N * DS; ad.osI = 0; ad.osQ = DS;
    k_attn<32><<<dim3(3, (unsigned)(B * H)), 128, 0, stream>>>(ad);
  }
  gemm(xs, F(12), F(13), sg, (int)BN, DS, DS);            // gate(x)
  gemm(so, F(14), F(15), sol, (int)BN, DS, DS);           // out(o)
  float* out_s = (float*)d_out;
  hipMemcpyAsync(out_s, s_in, SN * sizeof(float), hipMemcpyDeviceToDevice, stream);
  k_resid_gate<<<ewg(SN), 256, 0, stream>>>(out_s, sg, sol, B, N, 1, DS, 0);
  hipMemcpyAsync(out_s + SN, zcur, ZE * sizeof(float), hipMemcpyDeviceToDevice, stream);
}
